// Block_66245575573745
// MI455X (gfx1250) — matrix operand order repaired and checked by reading, not compiled, not run on hardware
//
#include <hip/hip_runtime.h>
#include <hip/hip_bf16.h>
#include <math.h>

#define BB   2048
#define JJ   17
#define CC   480
#define KKA  3
#define TOKN 64
#define CHN  960
#define HH   6
#define HDIM 80
#define EPSV 1e-5f

typedef __attribute__((ext_vector_type(16))) _Float16 v16h;
typedef __attribute__((ext_vector_type(8)))  _Float16 v8h;
typedef __attribute__((ext_vector_type(8)))  float    v8f;

// Types matching the async-load builtin prototype (b128 = int x4 vector),
// in explicit address spaces (AS1 = global, AS3 = LDS).
typedef int v4i_ __attribute__((vector_size(16)));
typedef __attribute__((address_space(1))) v4i_ as1_v4i;
typedef __attribute__((address_space(3))) v4i_ as3_v4i;

#if defined(__has_builtin)
#if __has_builtin(__builtin_amdgcn_global_load_async_to_lds_b128) && \
    __has_builtin(__builtin_amdgcn_s_wait_asynccnt)
#define USE_ASYNC_LDS 1
#endif
#endif

__device__ __forceinline__ float gelu_f(float x) {
    return 0.5f * x * (1.0f + erff(x * 0.70710678118654752f));
}

// ---------------------------------------------------------------------------
// WMMA GEMM:  C[M,N] = act(A[M,K] * W[N,K]^T + bias[N])
// A: f16 row-major. W: f16 row-major N x K (i.e. B^T).
// Block = 8 waves = 128(M) x 96(N) tile; each wave owns a 16x96 slice.
// B tile (96x32 per K-step) is staged once per block into LDS, double-
// buffered via GLOBAL_LOAD_ASYNC_TO_LDS_B128 (ASYNCcnt). Per-thread
// staging addresses are precomputed once; stage() is straight-line
// (chunk address advances by exactly k0 halves per step).
// LDS layout = fragment order: chunk (t,lane) = the 16 contiguous halves
// lane needs for v_wmma tile t, so compute is one aligned 32B ds read.
// M%128==0, N%96==0, K%32==0 (exact here; EXEC all-ones around WMMA).
// ---------------------------------------------------------------------------
__global__ __launch_bounds__(256) void k_gemm(
    const _Float16* __restrict__ A, const _Float16* __restrict__ W,
    const float* __restrict__ bias, float* __restrict__ Cf,
    _Float16* __restrict__ Ch, int M, int N, int K, int act, int outHalf)
{
    __shared__ __align__(32) _Float16 bs[2][3072];   // 2 x 6KB

    const int lane = threadIdx.x & 31;
    const int wid  = threadIdx.x >> 5;
    const int nStrips = N / 96;
    const int n0 = (blockIdx.x % nStrips) * 96;
    const int m0 = (blockIdx.x / nStrips) * 128 + wid * 16;

    // --- Per-thread staging descriptors (computed once) ---
    // 384 16B chunks per tile; thread tid owns chunk tid, and (tid<128)
    // additionally chunk tid+256. Chunk i -> tile t=i>>6, lane lc=(i&63)>>1,
    // 16B half h=i&1 of the 32B fragment.
    auto chunkPtr = [&](int i) -> const _Float16* {
        const int t  = i >> 6;
        const int r  = i & 63;
        const int lc = r >> 1;
        const int h  = r & 1;
        return W + (size_t)(n0 + t * 16 + (lc & 15)) * K + ((lc >> 4) << 3) + (h << 4);
    };
    const int i0 = threadIdx.x;
    const int i1 = (threadIdx.x + 256 < 384) ? threadIdx.x + 256 : 0;
    const bool has2 = threadIdx.x < 128;
    const _Float16* g0 = chunkPtr(i0);
    const _Float16* g1 = chunkPtr(i1);
    _Float16* l0 = &bs[0][i0 * 8];
    _Float16* l1 = &bs[0][i1 * 8];

    auto stage = [&](int buf, int k0) {
        const int bo = buf * 3072;
#ifdef USE_ASYNC_LDS
        __builtin_amdgcn_global_load_async_to_lds_b128(
            (as1_v4i*)(uintptr_t)(g0 + k0), (as3_v4i*)(uintptr_t)(l0 + bo), 0, 0);
        if (has2)
            __builtin_amdgcn_global_load_async_to_lds_b128(
                (as1_v4i*)(uintptr_t)(g1 + k0), (as3_v4i*)(uintptr_t)(l1 + bo), 0, 0);
#else
        *(v8h*)(l0 + bo) = *(const v8h*)(g0 + k0);
        if (has2) *(v8h*)(l1 + bo) = *(const v8h*)(g1 + k0);
#endif
    };

    v8f acc[6];
#pragma unroll
    for (int t = 0; t < 6; ++t) acc[t] = (v8f){0.f,0.f,0.f,0.f,0.f,0.f,0.f,0.f};

    // A fragment addressing (16-bit A 16x32 ISA layout):
    // lane<16 : halves 0-7 = K[k0+0..7],  halves 8-15 = K[k0+16..23]
    // lane>=16: halves 0-7 = K[k0+8..15], halves 8-15 = K[k0+24..31]
    const int rowA = m0 + (lane & 15);
    const _Float16* aBase = A + (size_t)rowA * K + ((lane >> 4) << 3);

    int buf = 0;
    stage(0, 0);
    for (int k0 = 0; k0 < K; k0 += 32) {
#ifdef USE_ASYNC_LDS
        __builtin_amdgcn_s_wait_asynccnt(0);
#endif
        __syncthreads();                       // bs[buf] ready for all waves
        if (k0 + 32 < K) stage(buf ^ 1, k0 + 32);

        v8h alo = *(const v8h*)(aBase + k0);
        v8h ahi = *(const v8h*)(aBase + k0 + 16);
        v16h a  = __builtin_shufflevector(alo, ahi,
                    0,1,2,3,4,5,6,7,8,9,10,11,12,13,14,15);
#pragma unroll
        for (int t = 0; t < 6; ++t) {
            v16h bfrag = *(const v16h*)(&bs[buf][(t * 32 + lane) << 4]);
            acc[t] = __builtin_amdgcn_wmma_f32_16x16x32_f16(
                         false, a, false, bfrag, (short)0, acc[t], false, false);
        }
        __syncthreads();                       // done reading bs[buf]
        buf ^= 1;
    }

#pragma unroll
    for (int t = 0; t < 6; ++t) {
        const int n = n0 + t * 16 + (lane & 15);
        const float bs_ = bias ? bias[n] : 0.f;
#pragma unroll
        for (int r = 0; r < 8; ++r) {
            const int m = m0 + r + ((lane >> 4) << 3);
            float v = acc[t][r] + bs_;
            if (act) v = gelu_f(v);
            if (outHalf) Ch[(size_t)m * N + n] = (_Float16)v;
            else         Cf[(size_t)m * N + n] = v;
        }
    }
}

// ---------------------------------------------------------------------------
__global__ void k_f2h(const float* __restrict__ s, _Float16* __restrict__ d, int n)
{
    int i = blockIdx.x * blockDim.x + threadIdx.x;
    if (i < n) d[i] = (_Float16)s[i];
}

// LN over joints (J=17). Writes XN f32 (B,C,J) and XN f16 (B*J, C).
__global__ void k_ln1(const float* __restrict__ x, const float* __restrict__ g,
                      const float* __restrict__ bta, float* __restrict__ XN,
                      _Float16* __restrict__ XNa)
{
    int idx = blockIdx.x * blockDim.x + threadIdx.x;   // b*C + c
    int b = idx / CC, c = idx % CC;
    const float* xp = x + (size_t)b * JJ * CC + c;
    float v[JJ]; float s = 0.f, s2 = 0.f;
#pragma unroll
    for (int j = 0; j < JJ; ++j) { float t = xp[(size_t)j * CC]; v[j] = t; s += t; s2 += t * t; }
    float mu  = s * (1.f / JJ);
    float var = s2 * (1.f / JJ) - mu * mu;
    float r   = rsqrtf(var + EPSV);
    float* xo = XN + (size_t)idx * JJ;
#pragma unroll
    for (int j = 0; j < JJ; ++j) {
        float t = (v[j] - mu) * r * g[j] + bta[j];
        xo[j] = t;
        XNa[((size_t)b * JJ + j) * CC + c] = (_Float16)t;
    }
}

// depthwise conv width 7, pad 3
__global__ void k_dw(const float* __restrict__ XN, const float* __restrict__ w,
                     const float* __restrict__ bias, float* __restrict__ out)
{
    int idx = blockIdx.x * blockDim.x + threadIdx.x;   // b*C + c
    int c = idx % CC;
    const float* xr = XN + (size_t)idx * JJ;
    float wv[7];
#pragma unroll
    for (int t = 0; t < 7; ++t) wv[t] = w[c * 7 + t];
    float bb = bias[c];
    float* o = out + (size_t)idx * JJ;
#pragma unroll
    for (int j = 0; j < JJ; ++j) {
        float a = bb;
#pragma unroll
        for (int t = 0; t < 7; ++t) {
            int p = j + t - 3;
            if (p >= 0 && p < JJ) a += xr[p] * wv[t];
        }
        o[j] = a;
    }
}

// GroupNorm(16) stats per (b, group): mean & rstd over 30*17 elements
__global__ __launch_bounds__(64) void k_gnstats(const float* __restrict__ src,
                                                float* __restrict__ stats)
{
    int bg = blockIdx.x; int b = bg >> 4; int g = bg & 15;
    const int GC = CC / 16;      // 30
    const int NE = GC * JJ;      // 510
    __shared__ float sh[64], sh2[64];
    float s = 0.f, s2 = 0.f;
    for (int i = threadIdx.x; i < NE; i += 64) {
        int ch = g * GC + i / JJ;
        int j  = i - (i / JJ) * JJ;
        float v = src[((size_t)b * CC + ch) * JJ + j];
        s += v; s2 += v * v;
    }
    sh[threadIdx.x] = s; sh2[threadIdx.x] = s2;
    __syncthreads();
    for (int o = 32; o > 0; o >>= 1) {
        if (threadIdx.x < o) { sh[threadIdx.x] += sh[threadIdx.x + o]; sh2[threadIdx.x] += sh2[threadIdx.x + o]; }
        __syncthreads();
    }
    if (threadIdx.x == 0) {
        float mu  = sh[0] / NE;
        float var = sh2[0] / NE - mu * mu;
        stats[bg * 2]     = mu;
        stats[bg * 2 + 1] = rsqrtf(var + EPSV);
    }
}

// GN-apply + GELU + token MLP (17 -> 64 -> 17) per (b,c) row
__global__ void k_token(const float* __restrict__ dc, const float* __restrict__ stats,
                        const float* __restrict__ gg, const float* __restrict__ gb,
                        const float* __restrict__ w1, const float* __restrict__ b1,
                        const float* __restrict__ w2, const float* __restrict__ b2,
                        float* __restrict__ out)
{
    int idx = blockIdx.x * blockDim.x + threadIdx.x;   // b*C + c
    int b = idx / CC, c = idx % CC;
    int g = c / (CC / 16);
    float mu = stats[(b * 16 + g) * 2], rs = stats[(b * 16 + g) * 2 + 1];
    float sc = gg[c] * rs, sb = gb[c] - mu * sc;
    const float* dr = dc + (size_t)idx * JJ;
    float d[JJ];
#pragma unroll
    for (int j = 0; j < JJ; ++j) d[j] = gelu_f(dr[j] * sc + sb);
    float o[JJ];
#pragma unroll
    for (int j = 0; j < JJ; ++j) o[j] = b2[j];
    for (int t = 0; t < TOKN; ++t) {
        float h = b1[t];
#pragma unroll
        for (int j = 0; j < JJ; ++j) h += d[j] * w1[t * JJ + j];
        float gh = gelu_f(h);
#pragma unroll
        for (int j = 0; j < JJ; ++j) o[j] += gh * w2[j * TOKN + t];
    }
    float* op = out + (size_t)idx * JJ;
#pragma unroll
    for (int j = 0; j < JJ; ++j) op[j] = o[j];
}

// 17x17 attention, one wave per (b, head); writes O f16 in (B*J, C) layout
__global__ __launch_bounds__(256) void k_attn(const float* __restrict__ QKV,
                                              _Float16* __restrict__ Oa)
{
    __shared__ float sc[8][JJ][JJ + 1];
    int lane = threadIdx.x & 31, wid = threadIdx.x >> 5;
    int pair = blockIdx.x * 8 + wid;
    int b = pair / HH, h = pair % HH;
    const float* base = QKV + (size_t)b * JJ * (3 * CC) + h * HDIM;
    const float* Kb = base + CC;
    const float* Vb = base + 2 * CC;
    const float scale = 0.1118033988749895f;   // 80^-0.5
    for (int q = 0; q < JJ; ++q) {
        if (lane < JJ) {
            float s = 0.f;
            const float* qp = base + (size_t)q * (3 * CC);
            const float* kp = Kb + (size_t)lane * (3 * CC);
            for (int d = 0; d < HDIM; ++d) s += qp[d] * kp[d];
            sc[wid][q][lane] = s * scale;
        }
    }
    __syncthreads();
    if (lane < JJ) {
        float mx = -1e30f;
        for (int k = 0; k < JJ; ++k) mx = fmaxf(mx, sc[wid][lane][k]);
        float sm = 0.f;
        for (int k = 0; k < JJ; ++k) { float e = __expf(sc[wid][lane][k] - mx); sc[wid][lane][k] = e; sm += e; }
        float inv = 1.f / sm;
        for (int k = 0; k < JJ; ++k) sc[wid][lane][k] *= inv;
    }
    __syncthreads();
    for (int q = 0; q < JJ; ++q) {
        for (int d = lane; d < HDIM; d += 32) {
            float s = 0.f;
            for (int k = 0; k < JJ; ++k) s += sc[wid][q][k] * Vb[(size_t)k * (3 * CC) + d];
            Oa[((size_t)b * JJ + q) * CC + h * HDIM + d] = (_Float16)s;
        }
    }
}

// pos dwconv(9,pad4) + adjacency einsum over Y1 + proj + mjem -> s (B,C,J)
__global__ __launch_bounds__(256) void k_fuse_pre(
    const float* __restrict__ XN, const float* __restrict__ posw,
    const float* __restrict__ posb, const float* __restrict__ Y1,
    const float* __restrict__ adj, const float* __restrict__ proj,
    const float* __restrict__ mjem, float* __restrict__ s)
{
    __shared__ float adjs[KKA * JJ * JJ];   // 867 floats
    for (int i = threadIdx.x; i < KKA * JJ * JJ; i += 256) adjs[i] = adj[i];
    __syncthreads();
    int idx = blockIdx.x * 256 + threadIdx.x;   // b*C + c
    int b = idx / CC, c = idx % CC;
    const float* xr = XN + (size_t)idx * JJ;
    float acc[JJ];
    float pb = posb[c];
#pragma unroll
    for (int j = 0; j < JJ; ++j) acc[j] = pb;
#pragma unroll
    for (int t = 0; t < 9; ++t) {
        float w = posw[c * 9 + t];
#pragma unroll
        for (int j = 0; j < JJ; ++j) {
            int p = j + t - 4;
            if (p >= 0 && p < JJ) acc[j] += xr[p] * w;
        }
    }
    const float* yb = Y1 + (size_t)b * JJ * (KKA * CC) + c;
    for (int k = 0; k < KKA; ++k) {
        for (int v = 0; v < JJ; ++v) {
            float yv = yb[(size_t)v * (KKA * CC) + k * CC];
            const float* ar = adjs + (k * JJ + v) * JJ;
#pragma unroll
            for (int j = 0; j < JJ; ++j) acc[j] += yv * ar[j];
        }
    }
    const float* mr = mjem + (size_t)idx * JJ;
    float* so = s + (size_t)idx * JJ;
#pragma unroll
    for (int j = 0; j < JJ; ++j)
        so[j] = acc[j] + proj[((size_t)b * JJ + j) * CC + c] + mr[j];
}

// xc2[b,j,c] = x[b,j,c] + gelu(GN(s))
__global__ void k_fuse_post(const float* __restrict__ x, const float* __restrict__ s,
                            const float* __restrict__ stats, const float* __restrict__ gg,
                            const float* __restrict__ gb, float* __restrict__ xc2)
{
    int idx = blockIdx.x * blockDim.x + threadIdx.x;   // b*C + c
    int b = idx / CC, c = idx % CC;
    int g = c / (CC / 16);
    float mu = stats[(b * 16 + g) * 2], rs = stats[(b * 16 + g) * 2 + 1];
    float scl = gg[c] * rs, sb = gb[c] - mu * scl;
    const float* sr = s + (size_t)idx * JJ;
#pragma unroll
    for (int j = 0; j < JJ; ++j) {
        float f   = gelu_f(sr[j] * scl + sb);
        float xin = x[((size_t)b * JJ + j) * CC + c];
        xc2[((size_t)b * JJ + j) * CC + c] = xin + f;
    }
}

// LN over channels (C=480), one block per (b,j) row; writes XN2 f16
__global__ __launch_bounds__(128) void k_ln2(const float* __restrict__ xc2,
                                             const float* __restrict__ g,
                                             const float* __restrict__ bta,
                                             _Float16* __restrict__ XN2)
{
    __shared__ float sh[128], sh2[128];
    int row = blockIdx.x;
    const float* xr = xc2 + (size_t)row * CC;
    float s = 0.f, s2 = 0.f;
    for (int c = threadIdx.x; c < CC; c += 128) { float v = xr[c]; s += v; s2 += v * v; }
    sh[threadIdx.x] = s; sh2[threadIdx.x] = s2;
    __syncthreads();
    for (int o = 64; o > 0; o >>= 1) {
        if (threadIdx.x < o) { sh[threadIdx.x] += sh[threadIdx.x + o]; sh2[threadIdx.x] += sh2[threadIdx.x + o]; }
        __syncthreads();
    }
    float mu = sh[0] / CC;
    float rs = rsqrtf(sh2[0] / CC - mu * mu + EPSV);
    for (int c = threadIdx.x; c < CC; c += 128) {
        float v = (xr[c] - mu) * rs * g[c] + bta[c];
        XN2[(size_t)row * CC + c] = (_Float16)v;
    }
}

// out[b,j,c] = xc2 + mlp_out + adjacency-einsum(Y2, adj)
__global__ __launch_bounds__(256) void k_final(const float* __restrict__ xc2,
                                               const float* __restrict__ mlp,
                                               const float* __restrict__ Y2,
                                               const float* __restrict__ adj,
                                               float* __restrict__ out)
{
    __shared__ float adjs[KKA * JJ * JJ];
    for (int i = threadIdx.x; i < KKA * JJ * JJ; i += 256) adjs[i] = adj[i];
    __syncthreads();
    size_t idx = (size_t)blockIdx.x * 256 + threadIdx.x;   // (b*J + j)*C + c
    int c = (int)(idx % CC);
    size_t bj = idx / CC;
    int j = (int)(bj % JJ);
    size_t b = bj / JJ;
    const float* yb = Y2 + ((size_t)b * JJ) * (KKA * CC) + c;
    float acc = 0.f;
    for (int k = 0; k < KKA; ++k)
        for (int v = 0; v < JJ; ++v)
            acc += yb[(size_t)v * (KKA * CC) + k * CC] * adjs[(k * JJ + v) * JJ + j];
    out[idx] = xc2[idx] + mlp[idx] + acc;
}

// ---------------------------------------------------------------------------
extern "C" void kernel_launch(void* const* d_in, const int* in_sizes, int n_in,
                              void* d_out, int out_size, void* d_ws, size_t ws_size,
                              hipStream_t stream)
{
    const float* x     = (const float*)d_in[0];
    const float* adj   = (const float*)d_in[1];
    const float* n1g   = (const float*)d_in[2];
    const float* n1b   = (const float*)d_in[3];
    const float* gcn1w = (const float*)d_in[4];
    const float* gcn1b = (const float*)d_in[5];
    const float* dww   = (const float*)d_in[6];
    const float* dwb   = (const float*)d_in[7];
    const float* dgng  = (const float*)d_in[8];
    const float* dgnb  = (const float*)d_in[9];
    const float* m1w1  = (const float*)d_in[10];
    const float* m1b1  = (const float*)d_in[11];
    const float* m1w2  = (const float*)d_in[12];
    const float* m1b2  = (const float*)d_in[13];
    const float* qkvw  = (const float*)d_in[14];
    const float* qkvb  = (const float*)d_in[15];
    const float* projw = (const float*)d_in[16];
    const float* projb = (const float*)d_in[17];
    const float* posw  = (const float*)d_in[18];
    const float* posb  = (const float*)d_in[19];
    const float* gnfg  = (const float*)d_in[20];
    const float* gnfb  = (const float*)d_in[21];
    const float* n2g   = (const float*)d_in[22];
    const float* n2b   = (const float*)d_in[23];
    const float* gcn2w = (const float*)d_in[24];
    const float* gcn2b = (const float*)d_in[25];
    const float* m2w1  = (const float*)d_in[26];
    const float* m2b1  = (const float*)d_in[27];
    const float* m2w2  = (const float*)d_in[28];
    const float* m2b2  = (const float*)d_in[29];

    char* ws = (char*)d_ws;
    const size_t nBCJ = (size_t)BB * CC * JJ;                 // 16,711,680
    const size_t nBJ  = (size_t)BB * JJ;                      // 34,816
    const size_t nBig = nBJ * (size_t)(KKA * CC);             // 50,135,040
    auto al = [](size_t v) { return (v + 255) & ~(size_t)255; };

    size_t off = 0;
    float*    XN   = (float*)(ws + off);    off += al(nBCJ * 4);
    _Float16* XNa  = (_Float16*)(ws + off); off += al(nBCJ * 2);
    float*    BIG1 = (float*)(ws + off);    off += al(nBig * 4);  // QKV -> Y2
    float*    BIG2 = (float*)(ws + off);    off += al(nBig * 4);  // Y1  -> Hmid(f16)
    float*    C1   = (float*)(ws + off);    off += al(nBCJ * 4);  // dconv -> s
    float*    C2   = (float*)(ws + off);    off += al(nBCJ * 4);  // mjem  -> xc2
    _Float16* HB   = (_Float16*)(ws + off); off += al(nBCJ * 2);  // o_a   -> XN2
    float*    PM   = (float*)(ws + off);    off += al(nBCJ * 4);  // proj  -> mlp_out
    float*    ST   = (float*)(ws + off);    off += al((size_t)BB * 16 * 2 * 4);
    _Float16* whGcn1 = (_Float16*)(ws + off); off += al((size_t)KKA * CC * CC * 2);
    _Float16* whQkv  = (_Float16*)(ws + off); off += al((size_t)3 * CC * CC * 2);
    _Float16* whProj = (_Float16*)(ws + off); off += al((size_t)CC * CC * 2);
    _Float16* whGcn2 = (_Float16*)(ws + off); off += al((size_t)KKA * CC * CC * 2);
    _Float16* whW1   = (_Float16*)(ws + off); off += al((size_t)CHN * CC * 2);
    _Float16* whW2   = (_Float16*)(ws + off); off += al((size_t)CC * CHN * 2);

    auto f2h = [&](const float* s, _Float16* d, int n) {
        k_f2h<<<(n + 255) / 256, 256, 0, stream>>>(s, d, n);
    };
    f2h(gcn1w, whGcn1, KKA * CC * CC);
    f2h(qkvw,  whQkv,  3 * CC * CC);
    f2h(projw, whProj, CC * CC);
    f2h(gcn2w, whGcn2, KKA * CC * CC);
    f2h(m2w1,  whW1,   CHN * CC);
    f2h(m2w2,  whW2,   CC * CHN);

    auto gemm = [&](const _Float16* A, const _Float16* W, const float* bias,
                    float* Cf, _Float16* Ch, int N, int K, int act, int oh) {
        const int M = BB * JJ;
        const int blocks = (M / 128) * (N / 96);
        k_gemm<<<blocks, 256, 0, stream>>>(A, W, bias, Cf, Ch, M, N, K, act, oh);
    };

    const int gBC = (BB * CC) / 256;   // 3840 blocks

    // --- spatial stage ---
    k_ln1<<<gBC, 256, 0, stream>>>(x, n1g, n1b, XN, XNa);
    gemm(XNa, whQkv, qkvb, BIG1, nullptr, 3 * CC, CC, 0, 0);           // QKV
    k_attn<<<(BB * HH) / 8, 256, 0, stream>>>(BIG1, HB);               // -> o_a (f16)
    gemm(HB, whProj, projb, PM, nullptr, CC, CC, 0, 0);                // proj
    gemm(XNa, whGcn1, gcn1b, BIG2, nullptr, KKA * CC, CC, 0, 0);       // Y1
    k_dw<<<gBC, 256, 0, stream>>>(XN, dww, dwb, C1);                   // dconv
    k_gnstats<<<BB * 16, 64, 0, stream>>>(C1, ST);
    k_token<<<gBC, 256, 0, stream>>>(C1, ST, dgng, dgnb, m1w1, m1b1, m1w2, m1b2, C2);
    k_fuse_pre<<<gBC, 256, 0, stream>>>(XN, posw, posb, BIG2, adj, PM, C2, C1);
    k_gnstats<<<BB * 16, 64, 0, stream>>>(C1, ST);
    k_fuse_post<<<gBC, 256, 0, stream>>>(x, C1, ST, gnfg, gnfb, C2);   // xc2 (b,j,c)

    // --- channel stage ---
    k_ln2<<<BB * JJ, 128, 0, stream>>>(C2, n2g, n2b, HB);              // XN2 (f16)
    gemm(HB, whGcn2, gcn2b, BIG1, nullptr, KKA * CC, CC, 0, 0);        // Y2
    gemm(HB, whW1, m2b1, nullptr, (_Float16*)BIG2, CHN, CC, 1, 1);     // Hmid = gelu(..), f16
    gemm((_Float16*)BIG2, whW2, m2b2, PM, nullptr, CC, CHN, 0, 0);     // mlp_out
    k_final<<<(int)(nBCJ / 256), 256, 0, stream>>>(C2, PM, BIG1, adj, (float*)d_out);
}
